// GCN_E_13692355740269
// MI455X (gfx1250) — matrix operand order repaired and checked by reading, not compiled, not run on hardware
//
#include <hip/hip_runtime.h>
#include <hip/hip_bf16.h>

typedef __attribute__((ext_vector_type(16))) __bf16 bf16x16;
typedef __attribute__((ext_vector_type(2)))  __bf16 bf16x2;
typedef __attribute__((ext_vector_type(8)))  float  f32x8;

// Pack two f32 into one dword of bf16 pairs; vector-init form pattern-matches
// to v_cvt_pk_bf16_f32 (RNE) on gfx1250.
static __device__ __forceinline__ unsigned pkbf16(float a, float b)
{
    const bf16x2 t = { (__bf16)a, (__bf16)b };
    return __builtin_bit_cast(unsigned, t);
}

// ---------------------------------------------------------------------------
// Tiled bf16 WMMA GEMM:  C[M x Ncols] = A[M x K] * B[K x Ncols]  (+ epilogue)
// Workgroup = 256 threads = 8 waves; block tile 64(M) x 64(N) x 32(K).
// Wave (wm 0..3, wn 0..1) owns a 16x32 C strip: 2 x v_wmma per k-step, A
// fragment reused. LDS tiles are stored in *fragment order* (ISA 7.12.2):
//   A: e = 8*(k>>4) + (k&7), lane = (m&15) + 16*((k>>3)&1)
//   B: e = 8*(k>>4) + (k&7),             lane = (n&15) + 16*((k>>3)&1)
// so each lane's 16-bf16 fragment is one contiguous 32B ds_load (b128 pair).
// EPI 0: store C as bf16 (support / s2 for L2-resident SpMM gathers)
// EPI 1: outF = relu(C + bias)            (attention hidden, f32)
// EPI 2: xio *= sigmoid(C + bias)         (attention gate applied in place)
// ---------------------------------------------------------------------------
template <int EPI>
__global__ __launch_bounds__(256)
void gemm_bf16_wmma(const float* __restrict__ A, const float* __restrict__ B,
                    const float* __restrict__ bias,
                    __bf16* __restrict__ outB, float* __restrict__ outF,
                    float* __restrict__ xio,
                    int M, int Ncols, int K)
{
    __shared__ __align__(32) __bf16 As[4 * 512];   // [wrow 0..3][lane 0..31][e 0..15]
    __shared__ __align__(32) __bf16 Bs[4 * 512];   // [ngrp 0..3][lane 0..31][e 0..15]

    const int tid  = threadIdx.x;
    const int wave = tid >> 5;
    const int lane = tid & 31;
    const int wm   = wave & 3;          // wave row 0..3
    const int wn   = wave >> 2;         // wave col 0..1 (each covers 32 cols)
    const int row0 = blockIdx.x * 64;
    const int col0 = blockIdx.y * 64;
    const int half = lane >> 4;
    const int l16  = lane & 15;

    f32x8 c0 = {}, c1 = {};

    // A stager: thread -> row ar (0..63), 8-aligned k-chunk ak (0/8/16/24).
    // Each 8-aligned chunk maps to 8 contiguous fragment elements of one lane.
    const int ar    = tid >> 2;
    const int ak    = (tid & 3) * 8;
    const int a_off = (ar >> 4) * 512
                    + ((ar & 15) + 16 * ((ak >> 3) & 1)) * 16
                    + 8 * (ak >> 4);
    const int gr    = row0 + ar;
    const float* __restrict__ arow_p = A + (size_t)gr * K + ak;

    // B stager: thread -> col bn (0..63), 8-aligned k-chunk bkc*8 (bkc 0..3).
    const int bn    = tid & 63;
    const int bkc   = tid >> 6;
    const int b_off = (bn >> 4) * 512
                    + ((bn & 15) + 16 * (bkc & 1)) * 16
                    + 8 * (bkc >> 1);

    for (int k0 = 0; k0 < K; k0 += 32) {
        { // stage A (coalesced f32x4 pair -> packed bf16 ds_store_b128)
            float v[8];
            if (gr < M) {
                const float4* p = (const float4*)(arow_p + k0);
                const float4 u0 = p[0], u1 = p[1];
                v[0]=u0.x; v[1]=u0.y; v[2]=u0.z; v[3]=u0.w;
                v[4]=u1.x; v[5]=u1.y; v[6]=u1.z; v[7]=u1.w;
                if (k0 + 32 < K)                       // hide HBM latency of the
                    __builtin_prefetch(arow_p + k0 + 32, 0, 0);  // streaming operand
            } else {
                #pragma unroll
                for (int i = 0; i < 8; ++i) v[i] = 0.f;
            }
            const uint4 w = { pkbf16(v[0], v[1]), pkbf16(v[2], v[3]),
                              pkbf16(v[4], v[5]), pkbf16(v[6], v[7]) };
            *(uint4*)&As[a_off] = w;
        }
        { // stage B: 8 k-strided dword loads (weights tiny + L2-resident),
          // one packed ds_store_b128
            const float* bp = B + (size_t)(k0 + bkc * 8) * Ncols + (col0 + bn);
            float v[8];
            #pragma unroll
            for (int i = 0; i < 8; ++i) v[i] = bp[(size_t)i * Ncols];
            const uint4 w = { pkbf16(v[0], v[1]), pkbf16(v[2], v[3]),
                              pkbf16(v[4], v[5]), pkbf16(v[6], v[7]) };
            *(uint4*)&Bs[b_off] = w;
        }
        __syncthreads();

        const bf16x16 a  = *(const bf16x16*)&As[wm * 512 + lane * 16];
        const bf16x16 b0 = *(const bf16x16*)&Bs[(wn * 2 + 0) * 512 + lane * 16];
        const bf16x16 b1 = *(const bf16x16*)&Bs[(wn * 2 + 1) * 512 + lane * 16];
        c0 = __builtin_amdgcn_wmma_f32_16x16x32_bf16(false, a, false, b0,
                                                     (short)0, c0, false, false);
        c1 = __builtin_amdgcn_wmma_f32_16x16x32_bf16(false, a, false, b1,
                                                     (short)0, c1, false, false);
        __syncthreads();
    }

    // Epilogue. C layout: VGPR j -> M = j + 8*half, N = lane%16.
    // Strength-reduced addressing: one base index, idx += Ncols per row.
    #pragma unroll
    for (int tt = 0; tt < 2; ++tt) {
        const f32x8 cc   = tt ? c1 : c0;
        const int   ccol = col0 + wn * 32 + tt * 16 + l16;
        const int   rbase = row0 + wm * 16 + 8 * half;
        const float bv   = (EPI != 0) ? bias[ccol] : 0.f;
        size_t idx = (size_t)rbase * Ncols + ccol;
        #pragma unroll
        for (int j = 0; j < 8; ++j) {
            if (rbase + j < M) {
                float v = cc[j];
                if (EPI == 0) {
                    outB[idx] = (__bf16)v;
                } else if (EPI == 1) {
                    v += bv;
                    outF[idx] = v > 0.f ? v : 0.f;
                } else { // EPI == 2
                    v += bv;
                    const float s = 1.f / (1.f + __expf(-v));
                    xio[idx] = xio[idx] * s;
                }
            }
            idx += (size_t)Ncols;
        }
    }
}

// ---------------------------------------------------------------------------
// SpMM scatter: one wave32 per edge. Gather bf16 feature row of col (L2-
// resident: 102 MB / 51 MB < 192 MB L2), scale by val, fp32-atomic into row.
// ---------------------------------------------------------------------------
__global__ __launch_bounds__(256)
void spmm_atomic(const int* __restrict__ rows, const int* __restrict__ cols,
                 const float* __restrict__ vals, const __bf16* __restrict__ S,
                 float* __restrict__ agg, int F, int E)
{
    const int gid  = blockIdx.x * blockDim.x + threadIdx.x;
    const int e    = gid >> 5;
    const int lane = gid & 31;
    if (e >= E) return;                       // wave-uniform exit
    const int   r  = rows[e];
    const int   cc = cols[e];
    const float v  = vals[e];
    const unsigned* src = (const unsigned*)(S + (size_t)cc * F);
    float* dst = agg + (size_t)r * F;
    const int nw = F >> 1;                    // packed dwords (2 x bf16)
    for (int i = lane; i < nw; i += 32) {
        const unsigned u = src[i];
        const float lo = __uint_as_float(u << 16);
        const float hi = __uint_as_float(u & 0xffff0000u);
        unsafeAtomicAdd(&dst[2 * i + 0], v * lo);
        unsafeAtomicAdd(&dst[2 * i + 1], v * hi);
    }
}

// ---------------------------------------------------------------------------
// Fused: y = gelu(y + bias) in place, accumulate column sum / sum-of-squares.
// ---------------------------------------------------------------------------
__global__ __launch_bounds__(256)
void gelu_stats(float* __restrict__ y, const float* __restrict__ bias,
                float* __restrict__ sum, float* __restrict__ sumsq,
                int Nrows, int F, int rowsPerBlock)
{
    const int tid = threadIdx.x;
    const int r0  = blockIdx.x * rowsPerBlock;
    const int r1  = min(r0 + rowsPerBlock, Nrows);
    const int nph = F >> 8;                   // 512 -> 2 phases, 256 -> 1
    float ls[2] = {0.f, 0.f}, lq[2] = {0.f, 0.f};
    for (int r = r0; r < r1; ++r) {
        for (int p = 0; p < nph; ++p) {
            const int col = p * 256 + tid;
            const size_t idx = (size_t)r * F + col;
            const float x = y[idx] + bias[col];
            const float g = 0.5f * x * (1.f + erff(x * 0.70710678118654752f));
            y[idx] = g;
            ls[p] += g;
            lq[p] += g * g;
        }
    }
    for (int p = 0; p < nph; ++p) {
        const int col = p * 256 + tid;
        unsafeAtomicAdd(&sum[col],   ls[p]);
        unsafeAtomicAdd(&sumsq[col], lq[p]);
    }
}

__global__ void bn_finalize(const float* __restrict__ sum, const float* __restrict__ sumsq,
                            const float* __restrict__ gamma, const float* __restrict__ beta,
                            float* __restrict__ scale, float* __restrict__ shift,
                            int F, float invN)
{
    const int f = blockIdx.x * blockDim.x + threadIdx.x;
    if (f < F) {
        const float m   = sum[f] * invN;
        const float var = sumsq[f] * invN - m * m;     // biased variance
        const float sc  = gamma[f] * rsqrtf(var + 1e-5f);
        scale[f] = sc;
        shift[f] = beta[f] - m * sc;
    }
}

__global__ __launch_bounds__(256)
void bn_apply(const float* __restrict__ in, float* __restrict__ out,
              const float* __restrict__ scale, const float* __restrict__ shift,
              size_t total, int Fmask)
{
    const size_t i = (size_t)blockIdx.x * blockDim.x + threadIdx.x;
    if (i < total) {
        const int col = (int)(i & (size_t)Fmask);      // F is a power of two
        out[i] = in[i] * scale[col] + shift[col];
    }
}

__global__ __launch_bounds__(256)
void fill_zero(float* __restrict__ p, size_t n)
{
    const size_t i = (size_t)blockIdx.x * blockDim.x + threadIdx.x;
    if (i < n) p[i] = 0.f;
}

// ---------------------------------------------------------------------------
extern "C" void kernel_launch(void* const* d_in, const int* in_sizes, int n_in,
                              void* d_out, int out_size, void* d_ws, size_t ws_size,
                              hipStream_t stream)
{
    const float* x    = (const float*)d_in[0];
    const int*   arow = (const int*)  d_in[1];
    const int*   acol = (const int*)  d_in[2];
    const float* aval = (const float*)d_in[3];
    const float* W1   = (const float*)d_in[4];
    const float* b1   = (const float*)d_in[5];
    const float* g1   = (const float*)d_in[6];
    const float* be1  = (const float*)d_in[7];
    const float* Wa1  = (const float*)d_in[8];
    const float* ba1  = (const float*)d_in[9];
    const float* Wa2  = (const float*)d_in[10];
    const float* ba2  = (const float*)d_in[11];
    const float* W2   = (const float*)d_in[12];
    const float* b2   = (const float*)d_in[13];
    const float* g2   = (const float*)d_in[14];
    const float* be2  = (const float*)d_in[15];

    const int K  = 512, H = 512, O = 256, AH = 128;
    const int N  = in_sizes[0] / K;
    const int E  = in_sizes[1];

    // Workspace layout (~512 MB)
    char* ws = (char*)d_ws;
    __bf16* supp = (__bf16*)ws;  ws += (size_t)N * H * 2;   // support, bf16 (L2-resident)
    float*  agg1 = (float*) ws;  ws += (size_t)N * H * 4;   // agg -> y -> x1 (in place)
    float*  ah   = (float*) ws;  ws += (size_t)N * AH * 4;  // attn hidden (relu), f32
    __bf16* s2   = (__bf16*)ws;  ws += (size_t)N * O * 2;   // layer-2 support, bf16
    float*  agg2 = (float*) ws;  ws += (size_t)N * O * 4;   // layer-2 agg -> y2
    float*  sum1 = (float*) ws;  ws += H * 4;
    float*  sq1  = (float*) ws;  ws += H * 4;
    float*  sc1  = (float*) ws;  ws += H * 4;
    float*  sh1  = (float*) ws;  ws += H * 4;
    float*  sum2 = (float*) ws;  ws += O * 4;
    float*  sq2  = (float*) ws;  ws += O * 4;
    float*  sc2  = (float*) ws;  ws += O * 4;
    float*  sh2  = (float*) ws;  ws += O * 4;

    const dim3 blk(256);
    const size_t nH = (size_t)N * H, nO = (size_t)N * O;

    // zero accumulators + stats (stats region is contiguous: 4H + 4O floats)
    fill_zero<<<(unsigned)((nH + 255) / 256), blk, 0, stream>>>(agg1, nH);
    fill_zero<<<(unsigned)((nO + 255) / 256), blk, 0, stream>>>(agg2, nO);
    fill_zero<<<(unsigned)((4 * H + 4 * O + 255) / 256), blk, 0, stream>>>(sum1, (size_t)(4 * H + 4 * O));

    // Layer 1: support = x @ W1  (bf16 WMMA, bf16 store)
    gemm_bf16_wmma<0><<<dim3((N + 63) / 64, H / 64), blk, 0, stream>>>(
        x, W1, nullptr, supp, nullptr, nullptr, N, H, K);

    // SpMM 1: agg1 += val * support[col]  (fp32 atomics)
    {
        const long long thr = (long long)E * 32;
        spmm_atomic<<<(unsigned)((thr + 255) / 256), blk, 0, stream>>>(
            arow, acol, aval, supp, agg1, H, E);
    }

    // gelu(agg1 + b1) in place + BN stats, then finalize + apply
    gelu_stats<<<(N + 127) / 128, blk, 0, stream>>>(agg1, b1, sum1, sq1, N, H, 128);
    bn_finalize<<<(H + 255) / 256, blk, 0, stream>>>(sum1, sq1, g1, be1, sc1, sh1, H, 1.f / (float)N);
    bn_apply<<<(unsigned)((nH + 255) / 256), blk, 0, stream>>>(agg1, agg1, sc1, sh1, nH, H - 1);

    // Attention MLP: ah = relu(x1 @ Wa1 + ba1);  x1 *= sigmoid(ah @ Wa2 + ba2)
    gemm_bf16_wmma<1><<<dim3((N + 63) / 64, AH / 64), blk, 0, stream>>>(
        agg1, Wa1, ba1, nullptr, ah, nullptr, N, AH, K);
    gemm_bf16_wmma<2><<<dim3((N + 63) / 64, H / 64), blk, 0, stream>>>(
        ah, Wa2, ba2, nullptr, nullptr, agg1, N, H, AH);

    // Layer 2: s2 = x1 @ W2 (bf16 store), SpMM, gelu+BN -> d_out
    gemm_bf16_wmma<0><<<dim3((N + 63) / 64, O / 64), blk, 0, stream>>>(
        agg1, W2, nullptr, s2, nullptr, nullptr, N, O, K);
    {
        const long long thr = (long long)E * 32;
        spmm_atomic<<<(unsigned)((thr + 255) / 256), blk, 0, stream>>>(
            arow, acol, aval, s2, agg2, O, E);
    }
    gelu_stats<<<(N + 127) / 128, blk, 0, stream>>>(agg2, b2, sum2, sq2, N, O, 128);
    bn_finalize<<<1, blk, 0, stream>>>(sum2, sq2, g2, be2, sc2, sh2, O, 1.f / (float)N);
    bn_apply<<<(unsigned)((nO + 255) / 256), blk, 0, stream>>>(agg2, (float*)d_out, sc2, sh2, nO, O - 1);
}
